// GraphNet_24498493456646
// MI455X (gfx1250) — matrix operand order repaired and checked by reading, not compiled, not run on hardware
//
#include <hip/hip_runtime.h>

// ---------------------------------------------------------------------------
// CDNA5 (gfx1250) implementation of the GraphNet reference.
// All GEMM-shaped work (convs via implicit GEMM, linears, GAT matmuls) goes
// through v_wmma_f32_16x16x32_bf16 (f32 accumulate, bf16 operands packed in
// LDS). wave32 layouts per CDNA5 ISA section 7.12.2.
// LDS tiles are double-buffered: stage tile t+1 while computing tile t,
// one workgroup barrier per K-step.
// ---------------------------------------------------------------------------

typedef __attribute__((ext_vector_type(16))) __bf16 v16bf;
typedef __attribute__((ext_vector_type(8)))  float  v8f;

__device__ __forceinline__ unsigned int pack2_bf16(float a, float b) {
  unsigned int ua = __float_as_uint(a);
  unsigned int ub = __float_as_uint(b);
  ua += 0x7fffu + ((ua >> 16) & 1u);     // round-to-nearest-even
  ub += 0x7fffu + ((ub >> 16) & 1u);
  return (ua >> 16) | (ub & 0xffff0000u);
}

__device__ __forceinline__ float act_apply(float v, int act) {
  switch (act) {
    case 1: return v > 0.f ? v : 0.f;                 // relu
    case 2: return v > 0.f ? v : (__expf(v) - 1.f);   // elu
    case 3: return 1.f / (1.f + __expf(-3.f * v));    // sigmoid(3x)
    default: return v;
  }
}

// ---------------------------------------------------------------------------
// Generic batched GEMM: C[z] = act(A[z] @ B[z] + bias)
// A: M x K row-major (ldA), B: bt==0 -> K x N (ldB), bt==1 -> N x K (ldB)
// C: M x N row-major (ldC).  Block tile 32(M) x 64(N), 8 waves of 32.
// Double-buffered LDS staging.
// ---------------------------------------------------------------------------
__global__ __launch_bounds__(256) void gemm_wmma_kernel(
    const float* __restrict__ A, const float* __restrict__ Bm,
    const float* __restrict__ bias, float* __restrict__ C,
    int M, int N, int K, int ldA, int ldB, int ldC,
    long long sA, long long sB, long long sC, int bt, int act)
{
  __shared__ unsigned int Ap[2][32][16];   // [buf][m][kpair]
  __shared__ unsigned int Bp[2][16][64];   // [buf][kpair][n]

  const int z = blockIdx.z;
  const float* Ab = A + (long long)z * sA;
  const float* Bb = Bm + (long long)z * sB;
  float* Cb = C + (long long)z * sC;

  const int m0 = blockIdx.y * 32;
  const int n0 = blockIdx.x * 64;
  const int tid = threadIdx.x;
  const int lane = tid & 31;
  const int wid = tid >> 5;
  const int wm = wid >> 2;      // 0..1
  const int wn = wid & 3;       // 0..3

  // per-thread constant staging coordinates
  const int a_kp = tid & 15;          // A: k-pair index (constant)
  const int a_ml = tid >> 4;          // A: base row
  const int b_n  = tid & 63;          // B: column (constant)
  const int b_kp = tid >> 6;          // B: base k-pair

  auto stage = [&](int kb, int bufi) {
    // ---- A: 32x16 pairs, 2 per thread (branchless, clamped addresses)
    {
      const int k = kb + 2 * a_kp;
      const int kc0 = k     < K ? k     : K - 1;
      const int kc1 = k + 1 < K ? k + 1 : K - 1;
      #pragma unroll
      for (int e = 0; e < 2; ++e) {
        int ml = a_ml + e * 16;
        int m = m0 + ml;
        int mc = m < M ? m : M - 1;
        const float* row = Ab + (long long)mc * ldA;
        float a0 = row[kc0];
        float a1 = row[kc1];
        a0 = (m < M && k     < K) ? a0 : 0.f;
        a1 = (m < M && k + 1 < K) ? a1 : 0.f;
        Ap[bufi][ml][a_kp] = pack2_bf16(a0, a1);
        // prefetch two K-tiles ahead (double-buffer covers tile+1)
        __builtin_prefetch(row + (k + 64 < K ? k + 64 : kc0), 0, 3);
      }
    }
    // ---- B: 16x64 pairs, 4 per thread (branchless)
    {
      int nn = n0 + b_n;
      int nc = nn < N ? nn : N - 1;
      #pragma unroll
      for (int e = 0; e < 4; ++e) {
        int kp = b_kp + e * 4;
        int k = kb + 2 * kp;
        int kc0 = k     < K ? k     : K - 1;
        int kc1 = k + 1 < K ? k + 1 : K - 1;
        float b0, b1;
        if (bt) {   // uniform branch
          const float* row = Bb + (long long)nc * ldB;
          b0 = row[kc0];
          b1 = row[kc1];
        } else {
          b0 = Bb[(long long)kc0 * ldB + nc];
          b1 = Bb[(long long)kc1 * ldB + nc];
        }
        b0 = (nn < N && k     < K) ? b0 : 0.f;
        b1 = (nn < N && k + 1 < K) ? b1 : 0.f;
        Bp[bufi][kp][b_n] = pack2_bf16(b0, b1);
      }
    }
  };

  v8f acc = {};
  stage(0, 0);
  __syncthreads();

  int buf = 0;
  for (int k0 = 0; k0 < K; k0 += 32) {
    if (k0 + 32 < K) stage(k0 + 32, buf ^ 1);   // overlap with compute below

    // ---- fragments per ISA 16-bit A (16x32) / B (32x16) lane layouts
    union { v16bf v; unsigned int u[8]; } af, bfm;
    const int mrow = wm * 16 + (lane & 15);
    const int hi = lane >> 4;
    #pragma unroll
    for (int v = 0; v < 8; ++v)
      af.u[v] = Ap[buf][mrow][((v < 4) ? v : v + 4) + hi * 4];
    const int ncol = wn * 16 + (lane & 15);
    #pragma unroll
    for (int v = 0; v < 8; ++v)
      bfm.u[v] = Bp[buf][((v < 4) ? v : v + 4) + hi * 4][ncol];

    acc = __builtin_amdgcn_wmma_f32_16x16x32_bf16(false, af.v, false, bfm.v,
                                                  (short)0, acc, false, false);
    __syncthreads();
    buf ^= 1;
  }

  // ---- store per C layout: VGPR r -> row wm*16 + r + (lane>=16)*8, col lane&15
  const int nn = n0 + wn * 16 + (lane & 15);
  if (nn < N) {
    const float bv = bias ? bias[nn] : 0.f;
    #pragma unroll
    for (int r = 0; r < 8; ++r) {
      int mm = m0 + wm * 16 + r + (lane >> 4) * 8;
      if (mm < M)
        Cb[(long long)mm * ldC + nn] = act_apply(acc[r] + bv, act);
    }
  }
}

// ---------------------------------------------------------------------------
// Implicit-GEMM conv2d (stride 1, zero pad, NCHW x OIHW), fused bias+relu.
// GEMM view: M = B*H*W, N = Cout, K = Cin*KH*KW.  Tile 128(M) x 32(N);
// each wave computes two 16x16 M-subtiles sharing one B fragment
// (2 x v_wmma per K-step per wave).  Double-buffered LDS staging.
// ---------------------------------------------------------------------------
__global__ __launch_bounds__(256) void conv2d_wmma_kernel(
    const float* __restrict__ x, const float* __restrict__ w,
    const float* __restrict__ bias, float* __restrict__ y,
    int Bn, int Cin, int H, int W, int Cout, int KH, int KW, int PH, int PW)
{
  __shared__ unsigned int Ap[2][128][16];
  __shared__ unsigned int Bp[2][16][32];

  const int K = Cin * KH * KW;
  const int HW = H * W;
  const int KHW = KH * KW;
  const int M = Bn * HW;

  const int m0 = blockIdx.x * 128;
  const int n0 = blockIdx.y * 32;
  const int tid = threadIdx.x;
  const int lane = tid & 31;
  const int wid = tid >> 5;
  const int wm = wid & 3;       // 0..3  -> rows wm*32 .. wm*32+31
  const int wn = wid >> 2;      // 0..1  -> cols wn*16 .. wn*16+15

  // per-thread constant staging coordinates
  const int a_kp = tid & 15;    // A k-pair (constant across e-loop)
  const int a_ml = tid >> 4;    // A base row (0..15)
  const int b_nl = tid & 31;    // B column (constant)
  const int b_kp0 = tid >> 5;   // B base k-pair

  auto stage = [&](int kb, int bufi) {
    // ---- decode this thread's two k values once per K-step
    const int ka = kb + 2 * a_kp;
    const int kca = ka     < K ? ka     : K - 1;
    const int kcb = ka + 1 < K ? ka + 1 : K - 1;
    const int ci0 = kca / KHW;
    const int r0 = (kca - ci0 * KHW) / KW;
    const int s0 = kca - ci0 * KHW - r0 * KW;
    const int ci1 = kcb / KHW;
    const int r1 = (kcb - ci1 * KHW) / KW;
    const int s1 = kcb - ci1 * KHW - r1 * KW;
    const bool kv0 = ka < K, kv1 = ka + 1 < K;

    // ---- stage A (im2col, branchless): 128x16 pairs, 8 per thread
    for (int e = 0; e < 8; ++e) {
      int ml = a_ml + e * 16;
      int m = m0 + ml;
      int mc = m < M ? m : M - 1;
      int b = mc / HW; int rem = mc - b * HW;
      int oh = rem / W; int ow = rem - oh * W;
      const float* xb = x + (long long)b * Cin * HW;

      int ih0 = oh - PH + r0, iw0 = ow - PW + s0;
      int ihc0 = min(max(ih0, 0), H - 1), iwc0 = min(max(iw0, 0), W - 1);
      float v0 = xb[((long long)ci0 * H + ihc0) * W + iwc0];
      v0 = (m < M && kv0 && ih0 >= 0 && ih0 < H && iw0 >= 0 && iw0 < W) ? v0 : 0.f;

      int ih1 = oh - PH + r1, iw1 = ow - PW + s1;
      int ihc1 = min(max(ih1, 0), H - 1), iwc1 = min(max(iw1, 0), W - 1);
      float v1 = xb[((long long)ci1 * H + ihc1) * W + iwc1];
      v1 = (m < M && kv1 && ih1 >= 0 && ih1 < H && iw1 >= 0 && iw1 < W) ? v1 : 0.f;

      Ap[bufi][ml][a_kp] = pack2_bf16(v0, v1);
    }
    // ---- stage B (weights, OIHW == N x K row-major): 16x32 pairs, 2/thread
    {
      int nn = n0 + b_nl;
      int nc = nn < Cout ? nn : Cout - 1;
      const float* wr = w + (long long)nc * K;
      #pragma unroll
      for (int e = 0; e < 2; ++e) {
        int kp = b_kp0 + e * 8;
        int k = kb + 2 * kp;
        int kc0 = k     < K ? k     : K - 1;
        int kc1 = k + 1 < K ? k + 1 : K - 1;
        float b0 = wr[kc0];
        float b1 = wr[kc1];
        b0 = (nn < Cout && k     < K) ? b0 : 0.f;
        b1 = (nn < Cout && k + 1 < K) ? b1 : 0.f;
        Bp[bufi][kp][b_nl] = pack2_bf16(b0, b1);
      }
    }
  };

  v8f acc0 = {}, acc1 = {};
  stage(0, 0);
  __syncthreads();

  int buf = 0;
  for (int k0 = 0; k0 < K; k0 += 32) {
    if (k0 + 32 < K) stage(k0 + 32, buf ^ 1);   // overlap with compute below

    union { v16bf v; unsigned int u[8]; } af0, af1, bfm;
    const int hi = lane >> 4;
    const int mrow0 = wm * 32 + (lane & 15);
    const int mrow1 = mrow0 + 16;
    #pragma unroll
    for (int v = 0; v < 8; ++v) {
      int kp = ((v < 4) ? v : v + 4) + hi * 4;
      af0.u[v] = Ap[buf][mrow0][kp];
      af1.u[v] = Ap[buf][mrow1][kp];
    }
    const int ncol = wn * 16 + (lane & 15);
    #pragma unroll
    for (int v = 0; v < 8; ++v)
      bfm.u[v] = Bp[buf][((v < 4) ? v : v + 4) + hi * 4][ncol];

    acc0 = __builtin_amdgcn_wmma_f32_16x16x32_bf16(false, af0.v, false, bfm.v,
                                                   (short)0, acc0, false, false);
    acc1 = __builtin_amdgcn_wmma_f32_16x16x32_bf16(false, af1.v, false, bfm.v,
                                                   (short)0, acc1, false, false);
    __syncthreads();
    buf ^= 1;
  }

  const int nn = n0 + wn * 16 + (lane & 15);
  if (nn < Cout) {
    const float bv = bias[nn];
    #pragma unroll
    for (int r = 0; r < 8; ++r) {
      int mm = m0 + wm * 32 + r + (lane >> 4) * 8;
      if (mm < M) {
        int b = mm / HW; int rem = mm - b * HW;
        int oh = rem / W; int ow = rem - oh * W;
        float v = acc0[r] + bv;
        y[(((long long)b * Cout + nn) * H + oh) * W + ow] = v > 0.f ? v : 0.f;
      }
      int mm1 = mm + 16;
      if (mm1 < M) {
        int b = mm1 / HW; int rem = mm1 - b * HW;
        int oh = rem / W; int ow = rem - oh * W;
        float v = acc1[r] + bv;
        y[(((long long)b * Cout + nn) * H + oh) * W + ow] = v > 0.f ? v : 0.f;
      }
    }
  }
}

// ---------------------------------------------------------------------------
// 2x2 max pool, stride 2.  x: (BC, H, W) -> y: (BC, H/2, W/2)
// ---------------------------------------------------------------------------
__global__ void maxpool2_kernel(const float* __restrict__ x, float* __restrict__ y,
                                int BC, int H, int W)
{
  const int OH = H >> 1, OW = W >> 1;
  int i = blockIdx.x * blockDim.x + threadIdx.x;
  int total = BC * OH * OW;
  if (i >= total) return;
  int ow = i % OW; int oh = (i / OW) % OH; int bc = i / (OW * OH);
  const float* p = x + ((long long)bc * H + 2 * oh) * W + 2 * ow;
  y[i] = fmaxf(fmaxf(p[0], p[1]), fmaxf(p[W], p[W + 1]));
}

// ---------------------------------------------------------------------------
// SE layer: out = x * sigmoid(relu(mean_N(x) @ w1.T) @ w2.T)
// x: (B, Nn, F).  grid.x = B, block = roundup32(F) threads.
// ---------------------------------------------------------------------------
__global__ void se_kernel(const float* __restrict__ x, const float* __restrict__ w1,
                          const float* __restrict__ w2, float* __restrict__ out,
                          int Nn, int F, int F4)
{
  __shared__ float ych[128];
  __shared__ float tch[32];
  const int b = blockIdx.x;
  const int c = threadIdx.x;
  const float* xb = x + (long long)b * Nn * F;
  if (c < F) {
    float s = 0.f;
    for (int n = 0; n < Nn; ++n) s += xb[n * F + c];
    ych[c] = s / (float)Nn;
  }
  __syncthreads();
  if (c < F4) {
    float s = 0.f;
    const float* wr = w1 + (long long)c * F;
    for (int k = 0; k < F; ++k) s += ych[k] * wr[k];
    tch[c] = s > 0.f ? s : 0.f;
  }
  __syncthreads();
  if (c < F) {
    float s = 0.f;
    const float* wr = w2 + (long long)c * F4;
    for (int k = 0; k < F4; ++k) s += tch[k] * wr[k];
    float zc = 1.f / (1.f + __expf(-s));
    float* ob = out + (long long)b * Nn * F;
    for (int n = 0; n < Nn; ++n) ob[n * F + c] = xb[n * F + c] * zc;
  }
}

// ---------------------------------------------------------------------------
// GAT attention scores: att[b,i,j] = softmax_j( mask(lrelu(a1.h_i + a2.h_j)) )
// mask = dist(coord_i, coord_j)^2 < 4 else -9e15.  grid=(Nn,B), block=64.
// ---------------------------------------------------------------------------
__global__ void attn_softmax_kernel(const float* __restrict__ h,
                                    const float* __restrict__ avec,
                                    const float* __restrict__ coord,
                                    float* __restrict__ att, int Nn, int Fh)
{
  __shared__ float red[64];
  const int b = blockIdx.y;
  const int i = blockIdx.x;
  const int j = threadIdx.x;
  const float* hb = h + (long long)b * Nn * Fh;
  const float* cb = coord + (long long)b * Nn * 2;
  float e = -9e15f;
  if (j < Nn) {
    float f1 = 0.f, f2 = 0.f;
    for (int k = 0; k < Fh; ++k) {
      f1 += hb[i * Fh + k] * avec[k];
      f2 += hb[j * Fh + k] * avec[Fh + k];
    }
    float v = f1 + f2;
    v = v > 0.f ? v : 0.2f * v;
    float dx = cb[i * 2 + 0] - cb[j * 2 + 0];
    float dy = cb[i * 2 + 1] - cb[j * 2 + 1];
    e = (dx * dx + dy * dy < 4.0f) ? v : -9e15f;
  }
  red[threadIdx.x] = e;
  __syncthreads();
  float m = -3e38f;
  for (int k = 0; k < Nn; ++k) m = fmaxf(m, red[k]);
  float ex = (j < Nn) ? __expf(e - m) : 0.f;
  __syncthreads();
  red[threadIdx.x] = ex;
  __syncthreads();
  float s = 0.f;
  for (int k = 0; k < Nn; ++k) s += red[k];
  if (j < Nn) att[((long long)b * Nn + i) * Nn + j] = ex / s;
}

// ---------------------------------------------------------------------------
// graph_feat: bilinear sample h (B,C,H,W) at normalized coords (B,Nn,2);
// writes feat (B,Nn,C) and pixel coords (B,Nn,2).  grid=(Nn,B), block>=C.
// ---------------------------------------------------------------------------
__global__ void graph_feat_kernel(const float* __restrict__ x,
                                  const float* __restrict__ cn,
                                  float* __restrict__ feat,
                                  float* __restrict__ coord_out,
                                  int C, int H, int W, int Nn)
{
  const int b = blockIdx.y, n = blockIdx.x, c = threadIdx.x;
  const float cr = cn[((long long)b * Nn + n) * 2 + 0] * (float)(H - 1);
  const float cc = cn[((long long)b * Nn + n) * 2 + 1] * (float)(W - 1);
  if (c == 0) {
    coord_out[((long long)b * Nn + n) * 2 + 0] = cr;
    coord_out[((long long)b * Nn + n) * 2 + 1] = cc;
  }
  if (c >= C) return;
  float ch = fminf(fmaxf(cr, 0.f), (float)(H - 1));
  float cw = fminf(fmaxf(cc, 0.f), (float)(W - 1));
  int h0 = (int)floorf(ch); int w0 = (int)floorf(cw);
  int h1 = min(h0 + 1, H - 1); int w1 = min(w0 + 1, W - 1);
  float th = ch - (float)h0, tw = cw - (float)w0;
  const float* p = x + ((long long)b * C + c) * H * W;
  float v00 = p[h0 * W + w0], v01 = p[h0 * W + w1];
  float v10 = p[h1 * W + w0], v11 = p[h1 * W + w1];
  feat[((long long)b * Nn + n) * C + c] =
      v00 * (1.f - th) * (1.f - tw) + v01 * (1.f - th) * tw +
      v10 * th * (1.f - tw) + v11 * th * tw;
}

__global__ void add_kernel(const float* __restrict__ a, const float* __restrict__ b,
                           float* __restrict__ c, int n)
{
  int i = blockIdx.x * blockDim.x + threadIdx.x;
  if (i < n) c[i] = a[i] + b[i];
}

// ---------------------------------------------------------------------------
// Host orchestration
// ---------------------------------------------------------------------------
enum {
  IN_X = 0,
  IN_C1W, IN_C1B, IN_C2W, IN_C2B, IN_C3W, IN_C3B, IN_C4W, IN_C4B, IN_C5W, IN_C5B,
  IN_R3W, IN_R3B, IN_R4W, IN_R4B, IN_R5W, IN_R5B,
  IN_FC1W, IN_FC1B, IN_FC2W, IN_FC2B,
  IN_G3_C11W, IN_G3_C11B, IN_G3_C12W, IN_G3_C12B, IN_G3_C21W, IN_G3_C21B,
  IN_G3_C22W, IN_G3_C22B, IN_G3_FCW, IN_G3_FCB,
  IN_G4_C1W, IN_G4_C1B, IN_G4_C2W, IN_G4_C2B, IN_G4_FC1W, IN_G4_FC1B,
  IN_G4_FC2W, IN_G4_FC2B,
  IN_G5_C11W, IN_G5_C11B, IN_G5_C12W, IN_G5_C12B, IN_G5_C21W, IN_G5_C21B,
  IN_G5_C22W, IN_G5_C22B, IN_G5_FCW, IN_G5_FCB,
  IN_GAT3_SE1W1, IN_GAT3_SE1W2, IN_GAT3_HW, IN_GAT3_HA,
  IN_GAT3_SE2W1, IN_GAT3_SE2W2, IN_GAT3_OW, IN_GAT3_OA,
  IN_GAT4_SE1W1, IN_GAT4_SE1W2, IN_GAT4_HW, IN_GAT4_HA,
  IN_GAT4_SE2W1, IN_GAT4_SE2W2, IN_GAT4_OW, IN_GAT4_OA,
  IN_GAT5_SE1W1, IN_GAT5_SE1W2, IN_GAT5_HW, IN_GAT5_HA,
  IN_GAT5_SE2W1, IN_GAT5_SE2W2, IN_GAT5_OW, IN_GAT5_OA
};

extern "C" void kernel_launch(void* const* d_in, const int* in_sizes, int n_in,
                              void* d_out, int out_size, void* d_ws, size_t ws_size,
                              hipStream_t stream)
{
  (void)in_sizes; (void)n_in; (void)out_size; (void)ws_size;
  const int Bn = 32;
  float* ws = (float*)d_ws;
  float* out = (float*)d_out;
  #define IN(i) ((const float*)d_in[(i)])

  // workspace slabs (floats)
  float* P = ws;                                   // 25,165,824
  float* Q = ws + 25165824;                        // 12,582,912
  float* R = ws + 25165824 + 12582912;             //  2,097,152
  // R sub-offsets
  float* r_cn   = R + 0;
  float* r_gf   = R + 4096;
  float* r_xp   = R + 135168;
  float* r_hb   = R + 266240;
  float* r_att  = R + 528384;
  float* r_xh   = R + 659456;
  float* r_xh2  = R + 921600;
  float* r_ho   = R + 1183744;
  float* r_atto = R + 1314816;
  float* r_gout = R + 1445888;
  float* r_tmp  = R + 1576960;

  // d_out region offsets
  float* o_g3 = out + 0;       float* o_c3 = out + 65536;
  float* o_g4 = out + 69632;   float* o_c4 = out + 102400;
  float* o_g5 = out + 104448;  float* o_c5 = out + 120832;
  float* o_xf = out + 121856;  float* o_out = out + 130048;

  auto conv = [&](const float* xi, const float* wi, const float* bi, float* yo,
                  int Cin, int H, int W, int Cout, int KH, int KW, int PH, int PW) {
    int M = Bn * H * W;
    dim3 g((M + 127) / 128, (Cout + 31) / 32, 1);
    conv2d_wmma_kernel<<<g, 256, 0, stream>>>(xi, wi, bi, yo, Bn, Cin, H, W,
                                              Cout, KH, KW, PH, PW);
  };
  auto gemm = [&](const float* A, const float* Bm, const float* bias, float* C,
                  int M, int N, int K, int ldA, int ldB, int ldC,
                  long long sA, long long sB, long long sC, int bt, int act, int batch) {
    dim3 g((N + 63) / 64, (M + 31) / 32, batch);
    gemm_wmma_kernel<<<g, 256, 0, stream>>>(A, Bm, bias, C, M, N, K,
                                            ldA, ldB, ldC, sA, sB, sC, bt, act);
  };
  auto pool = [&](const float* xi, float* yo, int BC, int H, int W) {
    int total = BC * (H / 2) * (W / 2);
    maxpool2_kernel<<<(total + 255) / 256, 256, 0, stream>>>(xi, yo, BC, H, W);
  };
  auto run_gat = [&](const float* gf, const float* coord, int Nn, int F,
                     const float* s1w1, const float* s1w2,
                     const float* hW, const float* ha,
                     const float* s2w1, const float* s2w2,
                     const float* oW, const float* oa,
                     const float* reduW, const float* reduB, float* outp) {
    const int F4 = F / 4;
    const int blkF = ((F + 31) / 32) * 32;
    se_kernel<<<Bn, blkF, 0, stream>>>(gf, s1w1, s1w2, r_xp, Nn, F, F4);
    for (int hd = 0; hd < 8; ++hd) {
      float* hh = r_hb + (long long)hd * Bn * Nn * 16;
      gemm(r_xp, hW + (long long)hd * F * 16, nullptr, hh,
           Nn, 16, F, F, 16, 16, (long long)Nn * F, 0, (long long)Nn * 16, 0, 0, Bn);
      attn_softmax_kernel<<<dim3(Nn, Bn), 64, 0, stream>>>(hh, ha + hd * 32, coord,
                                                           r_att, Nn, 16);
      gemm(r_att, hh, nullptr, r_xh + hd * 16,
           Nn, 16, Nn, Nn, 16, 128,
           (long long)Nn * Nn, (long long)Nn * 16, (long long)Nn * 128, 0, 2, Bn);
    }
    se_kernel<<<Bn, 128, 0, stream>>>(r_xh, s2w1, s2w2, r_xh2, Nn, 128, 32);
    gemm(r_xh2, oW, nullptr, r_ho, Nn, F, 128, 128, F, F,
         (long long)Nn * 128, 0, (long long)Nn * F, 0, 0, Bn);
    attn_softmax_kernel<<<dim3(Nn, Bn), 64, 0, stream>>>(r_ho, oa, coord, r_atto, Nn, F);
    gemm(r_atto, r_ho, nullptr, r_gout, Nn, F, Nn, Nn, F, F,
         (long long)Nn * Nn, (long long)Nn * F, (long long)Nn * F, 0, 2, Bn);
    int total = Bn * Nn * F;
    add_kernel<<<(total + 255) / 256, 256, 0, stream>>>(r_gout, gf, r_tmp, total);
    gemm(r_tmp, reduW, reduB, outp, Nn, 32, F, F, F, 32,
         (long long)Nn * F, 0, (long long)Nn * 32, 1, 2, Bn);
  };

  // ---------------- trunk ----------------
  conv(IN(IN_X), IN(IN_C1W), IN(IN_C1B), P, 1, 128, 256, 24, 5, 9, 2, 4);
  pool(P, Q, Bn * 24, 128, 256);                              // (B,24,64,128)
  conv(Q, IN(IN_C2W), IN(IN_C2B), P, 24, 64, 128, 48, 5, 7, 2, 3);
  pool(P, Q, Bn * 48, 64, 128);                               // (B,48,32,64)
  conv(Q, IN(IN_C3W), IN(IN_C3B), P, 48, 32, 64, 64, 5, 5, 2, 2);  // h3 = P

  // ---------------- scale 3 ----------------
  {
    float* q0 = Q, *q1 = Q + 2097152, *q2 = Q + 3145728, *q3 = Q + 3670016;
    conv(P,  IN(IN_G3_C11W), IN(IN_G3_C11B), q0, 64, 32, 64, 32, 1, 1, 0, 0);
    conv(q0, IN(IN_G3_C12W), IN(IN_G3_C12B), q1, 32, 32, 64, 16, 3, 3, 1, 1);
    conv(q1, IN(IN_G3_C21W), IN(IN_G3_C21B), q2, 16, 32, 64,  8, 1, 1, 0, 0);
    conv(q2, IN(IN_G3_C22W), IN(IN_G3_C22B), q3,  8, 32, 64,  4, 3, 3, 1, 1);
    gemm(q3, IN(IN_G3_FCW), IN(IN_G3_FCB), r_cn, Bn, 128, 8192,
         8192, 8192, 128, 0, 0, 0, 1, 3, 1);
    graph_feat_kernel<<<dim3(64, Bn), 64, 0, stream>>>(P, r_cn, r_gf, o_c3,
                                                       64, 32, 64, 64);
    run_gat(r_gf, o_c3, 64, 64,
            IN(IN_GAT3_SE1W1), IN(IN_GAT3_SE1W2), IN(IN_GAT3_HW), IN(IN_GAT3_HA),
            IN(IN_GAT3_SE2W1), IN(IN_GAT3_SE2W2), IN(IN_GAT3_OW), IN(IN_GAT3_OA),
            IN(IN_R3W), IN(IN_R3B), o_g3);
  }

  pool(P, Q, Bn * 64, 32, 64);                                // (B,64,16,32)
  conv(Q, IN(IN_C4W), IN(IN_C4B), P, 64, 16, 32, 96, 5, 5, 2, 2);  // h4 = P

  // ---------------- scale 4 ----------------
  {
    float* q0 = Q, *q1 = Q + 393216, *q2 = Q + 589824, *q3 = Q + 638976, *q4 = Q + 663552;
    pool(P, q0, Bn * 96, 16, 32);                             // (B,96,8,16)
    conv(q0, IN(IN_G4_C1W), IN(IN_G4_C1B), q1, 96, 8, 16, 48, 5, 5, 2, 2);
    pool(q1, q2, Bn * 48, 8, 16);                             // (B,48,4,8)
    conv(q2, IN(IN_G4_C2W), IN(IN_G4_C2B), q3, 48, 4, 8, 24, 5, 5, 2, 2);
    gemm(q3, IN(IN_G4_FC1W), IN(IN_G4_FC1B), q4, Bn, 128, 768,
         768, 768, 128, 0, 0, 0, 1, 1, 1);
    gemm(q4, IN(IN_G4_FC2W), IN(IN_G4_FC2B), r_cn, Bn, 64, 128,
         128, 128, 64, 0, 0, 0, 1, 3, 1);
    graph_feat_kernel<<<dim3(32, Bn), 96, 0, stream>>>(P, r_cn, r_gf, o_c4,
                                                       96, 16, 32, 32);
    run_gat(r_gf, o_c4, 32, 96,
            IN(IN_GAT4_SE1W1), IN(IN_GAT4_SE1W2), IN(IN_GAT4_HW), IN(IN_GAT4_HA),
            IN(IN_GAT4_SE2W1), IN(IN_GAT4_SE2W2), IN(IN_GAT4_OW), IN(IN_GAT4_OA),
            IN(IN_R4W), IN(IN_R4B), o_g4);
  }

  pool(P, Q, Bn * 96, 16, 32);                                // (B,96,8,16)
  conv(Q, IN(IN_C5W), IN(IN_C5B), P, 96, 8, 16, 96, 5, 5, 2, 2);   // h5 = P

  // ---------------- scale 5 ----------------
  {
    float* q0 = Q, *q1 = Q + 196608, *q2 = Q + 294912, *q3 = Q + 344064;
    conv(P,  IN(IN_G5_C11W), IN(IN_G5_C11B), q0, 96, 8, 16, 48, 1, 1, 0, 0);
    conv(q0, IN(IN_G5_C12W), IN(IN_G5_C12B), q1, 48, 8, 16, 24, 3, 3, 1, 1);
    conv(q1, IN(IN_G5_C21W), IN(IN_G5_C21B), q2, 24, 8, 16, 12, 1, 1, 0, 0);
    conv(q2, IN(IN_G5_C22W), IN(IN_G5_C22B), q3, 12, 8, 16,  6, 3, 3, 1, 1);
    gemm(q3, IN(IN_G5_FCW), IN(IN_G5_FCB), r_cn, Bn, 32, 768,
         768, 768, 32, 0, 0, 0, 1, 3, 1);
    graph_feat_kernel<<<dim3(16, Bn), 96, 0, stream>>>(P, r_cn, r_gf, o_c5,
                                                       96, 8, 16, 16);
    run_gat(r_gf, o_c5, 16, 96,
            IN(IN_GAT5_SE1W1), IN(IN_GAT5_SE1W2), IN(IN_GAT5_HW), IN(IN_GAT5_HA),
            IN(IN_GAT5_SE2W1), IN(IN_GAT5_SE2W2), IN(IN_GAT5_OW), IN(IN_GAT5_OA),
            IN(IN_R5W), IN(IN_R5B), o_g5);
  }

  // ---------------- head ----------------
  pool(P, Q, Bn * 96, 8, 16);                                 // (B,96,4,8) -> 3072
  gemm(Q, IN(IN_FC1W), IN(IN_FC1B), o_xf, Bn, 256, 3072,
       3072, 3072, 256, 0, 0, 0, 1, 1, 1);
  gemm(o_xf, IN(IN_FC2W), IN(IN_FC2B), o_out, Bn, 2000, 256,
       256, 256, 2000, 0, 0, 0, 1, 0, 1);
  #undef IN
}
